// DglHGTConv_39367670235355
// MI455X (gfx1250) — matrix operand order repaired and checked by reading, not compiled, not run on hardware
//
#include <hip/hip_runtime.h>
#include <hip/hip_bf16.h>
#include <math.h>

#define N_NODES 50000
#define E_EDGES 500000
#define NHEAD   8
#define DKH     16
#define NTILES  (N_NODES / 16)   // 3125, exact
#define LDS_STRIDE 136           // 128 + 8 halfs pad: 68 dwords -> conflict-free bank quads

typedef _Float16 f16;
typedef __attribute__((ext_vector_type(16))) _Float16 v16h;
typedef __attribute__((ext_vector_type(8)))  _Float16 v8h;
typedef __attribute__((ext_vector_type(8)))  float    v8f;

// ---- monotone float<->uint encoding for atomic max over signed floats ----
__device__ __forceinline__ unsigned fenc(float f) {
    unsigned u = __float_as_uint(f);
    return (u & 0x80000000u) ? ~u : (u | 0x80000000u);
}
__device__ __forceinline__ float fdec(unsigned u) {
    unsigned v = (u & 0x80000000u) ? (u & 0x7FFFFFFFu) : ~u;
    return __uint_as_float(v);
}

// ---------------- utility kernels ----------------
__global__ void k_f32_to_f16(const float* __restrict__ x, f16* __restrict__ y, int n) {
    int i = blockIdx.x * blockDim.x + threadIdx.x;
    if (i < n) y[i] = (f16)x[i];
}

__global__ void k_zero_f32(float* __restrict__ p, int n) {
    int i = blockIdx.x * blockDim.x + threadIdx.x;
    if (i < n) p[i] = 0.0f;
}

__global__ void k_gelu_to_f16(const float* __restrict__ x, f16* __restrict__ y, int n) {
    int i = blockIdx.x * blockDim.x + threadIdx.x;
    if (i < n) {
        float v = x[i];
        y[i] = (f16)(0.5f * v * (1.0f + erff(v * 0.70710678118654752f)));
    }
}

// ---------------- fold relation transforms into projection weights ----------------
// blockIdx.y selects matrix: 0=WqT, 1=WaT, 2=Wk*att[0], 3=Wk*att[1], 4=Wv*msg[0], 5=Wv*msg[1]
__global__ void k_fold_weights(const float* __restrict__ Wq, const float* __restrict__ Wa,
                               const float* __restrict__ Wk, const float* __restrict__ Wv,
                               const float* __restrict__ bk, const float* __restrict__ bv,
                               const float* __restrict__ rel_att, const float* __restrict__ rel_msg,
                               f16* __restrict__ wqT, f16* __restrict__ waT,
                               f16* __restrict__ wkT, f16* __restrict__ wvT,
                               float* __restrict__ bk_eff, float* __restrict__ bv_eff) {
    int m = blockIdx.y;
    int i = blockIdx.x * blockDim.x + threadIdx.x;
    if (i >= 128 * 128) return;
    int o = i >> 7, k = i & 127;                 // o = output col, k = input dim
    if (m == 0) { wqT[o * 128 + k] = (f16)Wq[k * 128 + o]; return; }
    if (m == 1) { waT[o * 128 + k] = (f16)Wa[k * 128 + o]; return; }
    int r = (m == 3 || m == 5) ? 1 : 0;
    bool isK = (m < 4);
    const float* W = isK ? Wk : Wv;
    const float* A = isK ? rel_att : rel_msg;
    const float* b = isK ? bk : bv;
    f16*   out  = (isK ? wkT : wvT) + r * 128 * 128;
    float* beff = (isK ? bk_eff : bv_eff) + r * 128;
    int head = o >> 4, t = o & 15;
    const float* Ar = A + ((r * NHEAD + head) * DKH) * DKH + t;   // A[r][head][j][t], stride DKH over j
    float s = 0.0f;
    #pragma unroll
    for (int j = 0; j < DKH; ++j)
        s += W[k * 128 + head * 16 + j] * Ar[j * DKH];
    out[o * 128 + k] = (f16)s;
    if (k == 0) {
        float sb = 0.0f;
        #pragma unroll
        for (int j = 0; j < DKH; ++j) sb += b[head * 16 + j] * Ar[j * DKH];
        beff[o] = sb;
    }
}

// ---------------- WMMA GEMM core (LDS-staged weights) ----------------
// Stages BT[128][128] (f16, transposed weights) into padded LDS, then each wave
// computes a 16-row x 128-col tile: 4 K-chunks x 8 col-tiles of v_wmma_f32_16x16x32_f16.
struct GemmFrag {
    v8f  acc[8];
    int  tile, half, l16;
    bool active;
};

__device__ __forceinline__ void gemm_core(const f16* __restrict__ A,
                                          const f16* __restrict__ BT,
                                          const float* __restrict__ bias,
                                          f16* Bs, GemmFrag& fr) {
    // cooperative stage: 256 threads x 128B = 32KB
    {
        int t = threadIdx.x;
        int row = t >> 1, hlf = (t & 1) * 64;
        const v8h* sp = (const v8h*)(BT + row * 128 + hlf);
        v8h* dp = (v8h*)(Bs + row * LDS_STRIDE + hlf);
        #pragma unroll
        for (int j = 0; j < 8; ++j) dp[j] = sp[j];
    }
    __syncthreads();

    const int lane = threadIdx.x & 31;
    const int wave = threadIdx.x >> 5;
    int tile = blockIdx.x * 8 + wave;
    fr.active = (tile < NTILES);
    if (!fr.active) tile = 0;          // tail waves compute redundantly, skip stores
    fr.tile = tile;
    fr.half = lane >> 4;
    fr.l16  = lane & 15;
    const int row = tile * 16 + fr.l16;

    // preload all 4 A fragments (distinct registers -> loads issue up front)
    v16h av[4];
    #pragma unroll
    for (int kc = 0; kc < 4; ++kc) {
        const int kA = kc * 32 + fr.half * 8;     // K = kA+{0..7} and kA+{16..23}
        v8h a0 = *(const v8h*)(A + row * 128 + kA);
        v8h a1 = *(const v8h*)(A + row * 128 + kA + 16);
        #pragma unroll
        for (int q = 0; q < 8; ++q) { av[kc][q] = a0[q]; av[kc][q + 8] = a1[q]; }
    }

    #pragma unroll
    for (int ct = 0; ct < 8; ++ct) {
        float b = bias[ct * 16 + fr.l16];
        #pragma unroll
        for (int v = 0; v < 8; ++v) fr.acc[ct][v] = b;
    }

    #pragma unroll
    for (int kc = 0; kc < 4; ++kc) {
        const int kB = kc * 32 + fr.half * 8;     // K = kB+{0..7} and kB+{16..23} for col l16
        v16h bf[8];
        #pragma unroll
        for (int ct = 0; ct < 8; ++ct) {
            const f16* bp = Bs + (ct * 16 + fr.l16) * LDS_STRIDE + kB;
            v8h b0 = *(const v8h*)(bp);
            v8h b1 = *(const v8h*)(bp + 16);
            #pragma unroll
            for (int q = 0; q < 8; ++q) { bf[ct][q] = b0[q]; bf[ct][q + 8] = b1[q]; }
        }
        #pragma unroll
        for (int ct = 0; ct < 8; ++ct)
            fr.acc[ct] = __builtin_amdgcn_wmma_f32_16x16x32_f16(
                false, av[kc], false, bf[ct], (short)0, fr.acc[ct], false, false);
    }
}

__global__ void __launch_bounds__(256)
k_gemm_f16out(const f16* __restrict__ A, const f16* __restrict__ BT,
              const float* __restrict__ bias, f16* __restrict__ out) {
    __shared__ f16 Bs[128 * LDS_STRIDE];
    GemmFrag fr;
    gemm_core(A, BT, bias, Bs, fr);
    if (!fr.active) return;
    #pragma unroll
    for (int ct = 0; ct < 8; ++ct)
        #pragma unroll
        for (int v = 0; v < 8; ++v)
            out[(fr.tile * 16 + fr.half * 8 + v) * 128 + ct * 16 + fr.l16] = (f16)fr.acc[ct][v];
}

__global__ void __launch_bounds__(256)
k_gemm_final(const f16* __restrict__ A, const f16* __restrict__ BT,
             const float* __restrict__ bias, const float* __restrict__ resid,
             const float* __restrict__ skip, float* __restrict__ out) {
    __shared__ f16 Bs[128 * LDS_STRIDE];
    GemmFrag fr;
    gemm_core(A, BT, bias, Bs, fr);
    if (!fr.active) return;
    const float alpha = 1.0f / (1.0f + __expf(-skip[0]));
    const float beta  = 1.0f - alpha;
    #pragma unroll
    for (int ct = 0; ct < 8; ++ct)
        #pragma unroll
        for (int v = 0; v < 8; ++v) {
            int n = fr.tile * 16 + fr.half * 8 + v;
            int o = ct * 16 + fr.l16;
            out[n * 128 + o] = fr.acc[ct][v] * alpha + resid[n * 128 + o] * beta;
        }
}

// ---------------- edge phase ----------------
// one thread per (edge, head); consecutive threads = consecutive heads -> coalesced 32B chunks
__global__ void k_sddmm(const f16* __restrict__ q, const f16* __restrict__ kr,
                        const int* __restrict__ src, const int* __restrict__ dst,
                        const float* __restrict__ pri,
                        float* __restrict__ scores, unsigned* __restrict__ maxbuf) {
    int i = blockIdx.x * blockDim.x + threadIdx.x;
    if (i >= E_EDGES * NHEAD) return;
    int e = i >> 3, h = i & 7;
    int s = src[e], d = dst[e];
    v16h qv = *(const v16h*)(q  + d * 128 + h * 16);
    v16h kv = *(const v16h*)(kr + s * 128 + h * 16);
    float acc = 0.0f;
    #pragma unroll
    for (int j = 0; j < 16; ++j) acc += (float)qv[j] * (float)kv[j];
    float sc = acc * pri[h] * 0.25f;          // 1/sqrt(DK) = 0.25
    scores[i] = sc;
    atomicMax(maxbuf + d * NHEAD + h, fenc(sc));
}

__global__ void k_exp_sum(const int* __restrict__ dst,
                          float* __restrict__ scores,
                          const unsigned* __restrict__ maxbuf,
                          float* __restrict__ sumbuf) {
    int i = blockIdx.x * blockDim.x + threadIdx.x;
    if (i >= E_EDGES * NHEAD) return;
    int e = i >> 3, h = i & 7;
    int d = dst[e];
    float m  = fdec(maxbuf[d * NHEAD + h]);
    float ex = __expf(scores[i] - m);
    scores[i] = ex;
    atomicAdd(sumbuf + d * NHEAD + h, ex);
}

__global__ void k_aggregate(const f16* __restrict__ vr,
                            const int* __restrict__ src, const int* __restrict__ dst,
                            const float* __restrict__ scores,
                            const float* __restrict__ sumbuf,
                            float* __restrict__ tagg) {
    int i = blockIdx.x * blockDim.x + threadIdx.x;
    if (i >= E_EDGES * NHEAD) return;
    int e = i >> 3, h = i & 7;
    int s = src[e], d = dst[e];
    float attn = scores[i] / sumbuf[d * NHEAD + h];
    v16h vv = *(const v16h*)(vr + s * 128 + h * 16);
    float* tp = tagg + d * 128 + h * 16;
    #pragma unroll
    for (int j = 0; j < 16; ++j) atomicAdd(tp + j, (float)vv[j] * attn);
}

// ---------------- host launcher ----------------
extern "C" void kernel_launch(void* const* d_in, const int* in_sizes, int n_in,
                              void* d_out, int out_size, void* d_ws, size_t ws_size,
                              hipStream_t stream) {
    const float* h    = (const float*)d_in[0];
    const int* src[2] = { (const int*)d_in[1], (const int*)d_in[3] };
    const int* dst[2] = { (const int*)d_in[2], (const int*)d_in[4] };
    const float* Wk   = (const float*)d_in[5];
    const float* bk   = (const float*)d_in[6];
    const float* Wq   = (const float*)d_in[7];
    const float* bq   = (const float*)d_in[8];
    const float* Wv   = (const float*)d_in[9];
    const float* bv   = (const float*)d_in[10];
    const float* Wa   = (const float*)d_in[11];
    const float* ba   = (const float*)d_in[12];
    const float* ratt = (const float*)d_in[13];
    const float* rmsg = (const float*)d_in[14];
    const float* rpri = (const float*)d_in[15];
    const float* skip = (const float*)d_in[16];
    float* out = (float*)d_out;

    // workspace carve-up
    char* base = (char*)d_ws;
    size_t off = 0;
    auto carve = [&](size_t bytes) -> char* {
        char* p = base + off;
        off = (off + bytes + 255) & ~(size_t)255;
        return p;
    };
    f16*   h16    = (f16*)  carve((size_t)N_NODES * 128 * 2);   // later reused as t16
    f16*   wqT    = (f16*)  carve(128 * 128 * 2);
    f16*   waT    = (f16*)  carve(128 * 128 * 2);
    f16*   wkT    = (f16*)  carve(2 * 128 * 128 * 2);
    f16*   wvT    = (f16*)  carve(2 * 128 * 128 * 2);
    float* bk_eff = (float*)carve(2 * 128 * 4);
    float* bv_eff = (float*)carve(2 * 128 * 4);
    f16*   q16    = (f16*)  carve((size_t)N_NODES * 128 * 2);
    f16*   kr16   = (f16*)  carve((size_t)N_NODES * 128 * 2);
    f16*   vr16   = (f16*)  carve((size_t)N_NODES * 128 * 2);
    float* scores = (float*)carve((size_t)E_EDGES * NHEAD * 4);
    // contiguous zero region: t_agg + max/sum per relation
    float*    tagg   = (float*)   carve((size_t)N_NODES * 128 * 4);
    unsigned* maxb0  = (unsigned*)carve((size_t)N_NODES * NHEAD * 4);
    float*    sumb0  = (float*)   carve((size_t)N_NODES * NHEAD * 4);
    unsigned* maxb1  = (unsigned*)carve((size_t)N_NODES * NHEAD * 4);
    float*    sumb1  = (float*)   carve((size_t)N_NODES * NHEAD * 4);
    unsigned* maxb[2] = { maxb0, maxb1 };
    float*    sumb[2] = { sumb0, sumb1 };
    (void)ws_size; (void)n_in; (void)in_sizes; (void)out_size;

    const int TB = 256;
    const int nElem  = N_NODES * 128;                 // 6,400,000
    const int nEdgeT = E_EDGES * NHEAD;               // 4,000,000
    const int gemmBlocks = (NTILES + 7) / 8;          // 391

    // zero t_agg + max/sum buffers (bit-zero valid for f32 0.0 and the u32 max encoding)
    {
        int nz = nElem + 4 * N_NODES * NHEAD;         // 8,000,000 elements
        k_zero_f32<<<(nz + TB - 1) / TB, TB, 0, stream>>>(tagg, nz);
    }
    // h -> f16
    k_f32_to_f16<<<(nElem + TB - 1) / TB, TB, 0, stream>>>(h, h16, nElem);
    // fold weights (transposed f16) + effective biases
    {
        dim3 g((128 * 128 + TB - 1) / TB, 6);
        k_fold_weights<<<g, TB, 0, stream>>>(Wq, Wa, Wk, Wv, bk, bv, ratt, rmsg,
                                             wqT, waT, wkT, wvT, bk_eff, bv_eff);
    }
    // q projection
    k_gemm_f16out<<<gemmBlocks, TB, 0, stream>>>(h16, wqT, bq, q16);

    for (int r = 0; r < 2; ++r) {
        k_gemm_f16out<<<gemmBlocks, TB, 0, stream>>>(h16, wkT + r * 128 * 128, bk_eff + r * 128, kr16);
        k_gemm_f16out<<<gemmBlocks, TB, 0, stream>>>(h16, wvT + r * 128 * 128, bv_eff + r * 128, vr16);
        k_sddmm<<<(nEdgeT + TB - 1) / TB, TB, 0, stream>>>(q16, kr16, src[r], dst[r],
                                                           rpri + r * NHEAD, scores, maxb[r]);
        k_exp_sum<<<(nEdgeT + TB - 1) / TB, TB, 0, stream>>>(dst[r], scores, maxb[r], sumb[r]);
        k_aggregate<<<(nEdgeT + TB - 1) / TB, TB, 0, stream>>>(vr16, src[r], dst[r],
                                                               scores, sumb[r], tagg);
    }

    // GELU (exact, erf) -> f16, reusing h16 storage (h16 no longer needed)
    f16* t16 = h16;
    k_gelu_to_f16<<<(nElem + TB - 1) / TB, TB, 0, stream>>>(tagg, t16, nElem);

    // final projection + skip blend
    k_gemm_final<<<gemmBlocks, TB, 0, stream>>>(t16, waT, ba, h, skip, out);
}
